// MoE_18640158065014
// MI455X (gfx1250) — compile-verified, matrix operand order checked by reading, not run on hardware
//
#include <hip/hip_runtime.h>

#define D_MODEL 768
#define HIDDEN  2048
#define NEXP    8
#define NTOK    4096
#define ALPHA   0.05f

typedef __attribute__((ext_vector_type(16))) __bf16        v16bf;
typedef __attribute__((ext_vector_type(8)))  float         v8f;
typedef __attribute__((ext_vector_type(4)))  unsigned int  v4u;
typedef __attribute__((ext_vector_type(4)))  float         v4f;
typedef unsigned int u32;
typedef unsigned short u16;

union Frag { v4u q[2]; v16bf v; };

__device__ __forceinline__ u16 f2bf(float x) {
    union { float f; u32 u; } c; c.f = x;
    u32 r = (c.u + 0x7fffu + ((c.u >> 16) & 1u)) >> 16;
    return (u16)r;
}
__device__ __forceinline__ u32 pack2(float a, float b) {
    return (u32)f2bf(a) | ((u32)f2bf(b) << 16);
}
// 16x32 bf16 WMMA fragment: per-lane two contiguous 8-half runs,
// run0 at p[0..7], run1 at p[16..23] (halves). Caller applies kb offset.
__device__ __forceinline__ v16bf load_frag(const u16* p) {
    Frag f;
    f.q[0] = *(const v4u*)(p);
    f.q[1] = *(const v4u*)(p + 16);
    return f.v;
}

// ---------------------------------------------------------------- init
__global__ void init_ws_kernel(int* hdr) {
    if (threadIdx.x < 24) hdr[threadIdx.x] = 0;   // cnt[8], cnt2[8], spare
}

// ---------------------------------------------------------------- gate
__global__ void gate_kernel(const float* __restrict__ x,
                            const float* __restrict__ wgw,
                            const float* __restrict__ wgb,
                            int* __restrict__ cnt,
                            int* __restrict__ top1,
                            float* __restrict__ gate,
                            float* __restrict__ probs) {
    int lane = threadIdx.x & 31;
    int wave = threadIdx.x >> 5;
    int t = blockIdx.x * 8 + wave;
    if (t >= NTOK) return;
    const float* xr = x + (size_t)t * D_MODEL;
    float lg[NEXP];
#pragma unroll
    for (int e = 0; e < NEXP; e++) {
        const float* wr = wgw + e * D_MODEL;
        float s = 0.f;
        for (int d = lane; d < D_MODEL; d += 32) s += xr[d] * wr[d];
#pragma unroll
        for (int o = 16; o > 0; o >>= 1) s += __shfl_down(s, o, 32);
        lg[e] = s;
    }
    if (lane == 0) {
        float mx = -1e30f;
#pragma unroll
        for (int e = 0; e < NEXP; e++) { lg[e] += wgb[e]; mx = fmaxf(mx, lg[e]); }
        float p[NEXP], se = 0.f;
#pragma unroll
        for (int e = 0; e < NEXP; e++) { p[e] = __expf(lg[e] - mx); se += p[e]; }
        float inv = 1.f / se;
        int arg = 0; float best = lg[0];
#pragma unroll
        for (int e = 1; e < NEXP; e++) if (lg[e] > best) { best = lg[e]; arg = e; }
#pragma unroll
        for (int e = 0; e < NEXP; e++) probs[t * NEXP + e] = p[e] * inv;
        top1[t] = arg;
        gate[t] = p[arg] * inv;
        atomicAdd(&cnt[arg], 1);
    }
}

// ---------------------------------------------------------------- finalize
__global__ void finalize_kernel(const int* __restrict__ cnt,
                                const float* __restrict__ probs,
                                int* __restrict__ off,
                                float* __restrict__ auxp) {
    __shared__ float red[256];
    __shared__ float ces[NEXP];
    for (int e = 0; e < NEXP; e++) {
        float s = 0.f;
        for (int t = threadIdx.x; t < NTOK; t += 256) s += probs[t * NEXP + e];
        red[threadIdx.x] = s; __syncthreads();
        for (int st = 128; st > 0; st >>= 1) {
            if (threadIdx.x < st) red[threadIdx.x] += red[threadIdx.x + st];
            __syncthreads();
        }
        if (threadIdx.x == 0) ces[e] = red[0];
        __syncthreads();
    }
    if (threadIdx.x == 0) {
        int o = 0; float aux = 0.f;
#pragma unroll
        for (int e = 0; e < NEXP; e++) {
            off[e] = o; o += cnt[e];
            aux += ((float)cnt[e] / (float)NTOK) * (ces[e] / (float)NTOK);
        }
        *auxp = ALPHA * (float)NEXP * aux;
    }
}

// ---------------------------------------------------------------- scatter
__global__ void scatter_kernel(const int* __restrict__ top1,
                               const int* __restrict__ off,
                               int* __restrict__ cnt2,
                               int* __restrict__ perm) {
    int t = blockIdx.x * 256 + threadIdx.x;
    if (t >= NTOK) return;
    int e = top1[t];
    int pos = atomicAdd(&cnt2[e], 1);
    perm[off[e] + pos] = t;
}

// ---------------------------------------------------------------- weight cvt+T
// W: [NEXP][K][N] f32  ->  T: [NEXP][N][K] bf16
__global__ void convT_kernel(const float* __restrict__ W, u16* __restrict__ T,
                             int K, int N) {
    __shared__ u16 tile[64][65];
    int e = blockIdx.z;
    const float* We = W + (size_t)e * K * N;
    u16* Te = T + (size_t)e * K * N;
    int k0 = blockIdx.x * 64, n0 = blockIdx.y * 64;
    int tx = threadIdx.x & 63, ty = threadIdx.x >> 6;
    for (int i = ty; i < 64; i += 4)
        tile[i][tx] = f2bf(We[(size_t)(k0 + i) * N + n0 + tx]);
    __syncthreads();
    for (int i = ty; i < 64; i += 4)
        Te[(size_t)(n0 + i) * K + k0 + tx] = tile[tx][i];
}

// ---------------------------------------------------------------- GEMM1 fused
// H[p,:] = silu(Xe @ Wu) * (Xe @ Wv), expert-grouped rows.
// block: 128 thr (4 waves), M-tile=64 (4 WMMA M-subtiles), N=64 (16 cols/wave)
__global__ void __launch_bounds__(128)
gemm1_kernel(const float* __restrict__ x,
             const u16* __restrict__ WuT,
             const u16* __restrict__ WvT,
             const int* __restrict__ cnt,
             const int* __restrict__ off,
             const int* __restrict__ perm,
             u16* __restrict__ H) {
    int e  = blockIdx.x >> 6;
    int mt = blockIdx.x & 63;
    int ce = cnt[e];
    if (mt * 64 >= ce) return;                 // block-uniform exit
    int base = off[e] + mt * 64;

    __shared__ u16 As[64 * D_MODEL];           // 96 KB bf16 A-tile
    int tid = threadIdx.x;
    {
        int row = tid >> 1;                    // 0..63
        int seg = tid & 1;                     // 384 floats each
        u32* dst32 = (u32*)&As[row * D_MODEL + seg * 384];
        bool valid = (mt * 64 + row) < ce;
        if (valid) {
            int t = perm[base + row];
            const v4f* s4 = (const v4f*)(x + (size_t)t * D_MODEL + seg * 384);
#pragma unroll
            for (int j = 0; j < 96; j++) {
                v4f f = s4[j];
                dst32[j * 2 + 0] = pack2(f[0], f[1]);
                dst32[j * 2 + 1] = pack2(f[2], f[3]);
            }
        } else {
#pragma unroll
            for (int j = 0; j < 192; j++) dst32[j] = 0u;
        }
    }
    __syncthreads();

    int lane = tid & 31, wave = tid >> 5;
    int ncol = blockIdx.y * 64 + wave * 16 + (lane & 15);
    int kb = (lane >= 16) ? 8 : 0;
    const u16* ap0 = &As[(lane & 15) * D_MODEL];
    const u16* bu = WuT + ((size_t)e * HIDDEN + ncol) * D_MODEL;
    const u16* bv = WvT + ((size_t)e * HIDDEN + ncol) * D_MODEL;

    v8f zero = {0, 0, 0, 0, 0, 0, 0, 0};
    v8f uacc[4], vacc[4];
#pragma unroll
    for (int ms = 0; ms < 4; ms++) { uacc[ms] = zero; vacc[ms] = zero; }

    for (int k0 = 0; k0 < D_MODEL; k0 += 32) {
        v16bf fu = load_frag(bu + k0 + kb);
        v16bf fv = load_frag(bv + k0 + kb);
#pragma unroll
        for (int ms = 0; ms < 4; ms++) {
            v16bf a = load_frag(ap0 + ms * 16 * D_MODEL + k0 + kb);
            uacc[ms] = __builtin_amdgcn_wmma_f32_16x16x32_bf16(
                false, a, false, fu, (short)0, uacc[ms], false, false);
            vacc[ms] = __builtin_amdgcn_wmma_f32_16x16x32_bf16(
                false, a, false, fv, (short)0, vacc[ms], false, false);
        }
    }

    int mtop = (lane >= 16) ? 8 : 0;
#pragma unroll
    for (int ms = 0; ms < 4; ms++) {
#pragma unroll
        for (int r = 0; r < 8; r++) {
            int m = ms * 16 + r + mtop;
            if (mt * 64 + m < ce) {
                float u = uacc[ms][r], v = vacc[ms][r];
                float h = (u / (1.f + __expf(-u))) * v;   // silu(u)*v
                H[(size_t)(base + m) * HIDDEN + ncol] = f2bf(h);
            }
        }
    }
}

// ---------------------------------------------------------------- GEMM2
// y[t,:] = gate[t] * (H[p,:] @ Wd[e]).
// block: 128 thr (4 waves), M-tile=64, 2 N-tiles/wave (block N=128)
__global__ void __launch_bounds__(128)
gemm2_kernel(const u16* __restrict__ H,
             const u16* __restrict__ WdT,
             const int* __restrict__ cnt,
             const int* __restrict__ off,
             const int* __restrict__ perm,
             const float* __restrict__ gate,
             float* __restrict__ y) {
    int e  = blockIdx.x >> 6;
    int mt = blockIdx.x & 63;
    int ce = cnt[e];
    if (mt * 64 >= ce) return;
    int base = off[e] + mt * 64;

    int tid = threadIdx.x;
    int lane = tid & 31, wave = tid >> 5;
    int col0 = blockIdx.y * 128 + wave * 32 + (lane & 15);
    int kb = (lane >= 16) ? 8 : 0;

    const u16* ap[4];
#pragma unroll
    for (int ms = 0; ms < 4; ms++) {
        int mrow = base + ms * 16 + (lane & 15);
        if (mrow > NTOK - 1) mrow = NTOK - 1;  // padded rows masked at store
        ap[ms] = H + (size_t)mrow * HIDDEN;
    }
    const u16* bp0 = WdT + ((size_t)e * D_MODEL + col0) * HIDDEN;
    const u16* bp1 = bp0 + (size_t)16 * HIDDEN;

    v8f zero = {0, 0, 0, 0, 0, 0, 0, 0};
    v8f acc0[4], acc1[4];
#pragma unroll
    for (int ms = 0; ms < 4; ms++) { acc0[ms] = zero; acc1[ms] = zero; }

    for (int k0 = 0; k0 < HIDDEN; k0 += 32) {
        v16bf b0 = load_frag(bp0 + k0 + kb);
        v16bf b1 = load_frag(bp1 + k0 + kb);
#pragma unroll
        for (int ms = 0; ms < 4; ms++) {
            v16bf a = load_frag(ap[ms] + k0 + kb);
            acc0[ms] = __builtin_amdgcn_wmma_f32_16x16x32_bf16(
                false, a, false, b0, (short)0, acc0[ms], false, false);
            acc1[ms] = __builtin_amdgcn_wmma_f32_16x16x32_bf16(
                false, a, false, b1, (short)0, acc1[ms], false, false);
        }
    }

    int mtop = (lane >= 16) ? 8 : 0;
#pragma unroll
    for (int ms = 0; ms < 4; ms++) {
#pragma unroll
        for (int r = 0; r < 8; r++) {
            int m = ms * 16 + r + mtop;
            if (mt * 64 + m < ce) {
                int t = perm[base + m];
                float g = gate[t];
                y[(size_t)t * D_MODEL + col0]      = acc0[ms][r] * g;
                y[(size_t)t * D_MODEL + col0 + 16] = acc1[ms][r] * g;
            }
        }
    }
}

// ---------------------------------------------------------------- launch
extern "C" void kernel_launch(void* const* d_in, const int* in_sizes, int n_in,
                              void* d_out, int out_size, void* d_ws, size_t ws_size,
                              hipStream_t stream) {
    const float* x   = (const float*)d_in[0];
    const float* wgw = (const float*)d_in[1];
    const float* wgb = (const float*)d_in[2];
    const float* Wu  = (const float*)d_in[3];
    const float* Wv  = (const float*)d_in[4];
    const float* Wd  = (const float*)d_in[5];
    float* out = (float*)d_out;

    char* ws = (char*)d_ws;
    int*   cnt   = (int*)(ws + 0);          // 8 ints
    int*   cnt2  = (int*)(ws + 32);         // 8 ints
    int*   off   = (int*)(ws + 96);         // 8 ints
    int*   top1  = (int*)(ws + 128);                    // 4096 ints
    float* gate  = (float*)(ws + 128 + 16384);          // 4096 f32
    int*   perm  = (int*)(ws + 128 + 32768);            // 4096 ints
    float* probs = (float*)(ws + 128 + 49152);          // 4096*8 f32
    const size_t WB = 128 + 49152 + 131072;             // 180352, 16B aligned
    const size_t SZ = (size_t)NEXP * D_MODEL * HIDDEN * sizeof(u16);
    u16* WuT = (u16*)(ws + WB);
    u16* WvT = (u16*)(ws + WB + SZ);
    u16* WdT = (u16*)(ws + WB + 2 * SZ);
    u16* H   = (u16*)(ws + WB + 3 * SZ);    // 4096*2048 bf16
    float* auxp = out + (size_t)NTOK * D_MODEL;

    init_ws_kernel<<<1, 32, 0, stream>>>(cnt);

    convT_kernel<<<dim3(D_MODEL / 64, HIDDEN / 64, NEXP), 256, 0, stream>>>(Wu, WuT, D_MODEL, HIDDEN);
    convT_kernel<<<dim3(D_MODEL / 64, HIDDEN / 64, NEXP), 256, 0, stream>>>(Wv, WvT, D_MODEL, HIDDEN);
    convT_kernel<<<dim3(HIDDEN / 64, D_MODEL / 64, NEXP), 256, 0, stream>>>(Wd, WdT, HIDDEN, D_MODEL);

    gate_kernel<<<NTOK / 8, 256, 0, stream>>>(x, wgw, wgb, cnt, top1, gate, probs);
    finalize_kernel<<<1, 256, 0, stream>>>(cnt, probs, off, auxp);
    scatter_kernel<<<NTOK / 256, 256, 0, stream>>>(top1, off, cnt2, perm);

    gemm1_kernel<<<dim3(NEXP * 64, HIDDEN / 64), 128, 0, stream>>>(
        x, WuT, WvT, cnt, off, perm, H);
    gemm2_kernel<<<dim3(NEXP * 64, D_MODEL / 128), 128, 0, stream>>>(
        H, WdT, cnt, off, perm, gate, out);
}
